// CrossModalBlock_19164144075317
// MI455X (gfx1250) — matrix operand order repaired and checked by reading, not compiled, not run on hardware
//
#include <hip/hip_runtime.h>
#include <hip/hip_bf16.h>

// CrossModalBlock for MI455X (gfx1250, wave32, WMMA + TDM).
// All GEMMs via v_wmma_f32_16x16x32_bf16 (f32 accum); softmax/LN in f32.
// K-tiles staged to LDS with tensor_load_to_lds (TDM) in the attention kernel.
// GEMM k-loop: 2-stage unrolled software pipeline (disjoint register sets,
// no rotation moves, partial loadcnt waits overlap loads with WMMA).

typedef __attribute__((ext_vector_type(16))) __bf16        v16bf;
typedef __attribute__((ext_vector_type(8)))  float         v8f;
typedef __attribute__((ext_vector_type(8)))  unsigned int  v8u;
typedef __attribute__((ext_vector_type(4)))  unsigned int  u32x4;
typedef __attribute__((ext_vector_type(4)))  int           i32x4;
typedef __attribute__((ext_vector_type(8)))  int           i32x8;

#define DEV static __device__ __forceinline__

DEV unsigned short f2bf(float f) {
  unsigned u = __builtin_bit_cast(unsigned, f);
  u = (u + 0x7FFFu + ((u >> 16) & 1u)) >> 16;   // round-to-nearest-even
  return (unsigned short)u;
}
DEV unsigned pack2(float a, float b) {
  return (unsigned)f2bf(a) | ((unsigned)f2bf(b) << 16);
}

// A-fragment: 16x32 bf16 (MxK), row-major source, stride ld (elements).
// Lane l holds row M=l&15; lanes 0-15: K 0..7 & 16..23, lanes 16-31: K 8..15 & 24..31.
template <typename PT>
DEV v16bf fragA(PT base, int ld) {
  int lane = threadIdx.x & 31;
  auto p = (const unsigned*)(base + (size_t)(lane & 15) * ld + ((lane >> 4) << 3));
  v8u u;
#pragma unroll
  for (int i = 0; i < 4; ++i) { u[i] = p[i]; u[4 + i] = p[8 + i]; }
  return __builtin_bit_cast(v16bf, u);
}

// B-fragment: 32x16 bf16 (KxN) where B[k][n] = W[n][k], W row-major [N,K] stride ld.
// Lane l holds column N=l&15; lanes 0-15: K 0..7 & 16..23, lanes 16-31: K 8..15 & 24..31 (packed pairs).
template <typename PT>
DEV v16bf fragB(PT base, int ld) {
  int lane = threadIdx.x & 31;
  auto p = (const unsigned*)(base + (size_t)(lane & 15) * ld + ((lane >> 4) << 3));
  v8u u;
#pragma unroll
  for (int i = 0; i < 4; ++i) { u[i] = p[i]; u[4 + i] = p[8 + i]; }
  return __builtin_bit_cast(v16bf, u);
}

DEV float wred_max(float v) {
#pragma unroll
  for (int m = 16; m > 0; m >>= 1) { float o = __shfl_xor(v, m, 32); v = v > o ? v : o; }
  return v;
}
DEV float wred_sum(float v) {
#pragma unroll
  for (int m = 16; m > 0; m >>= 1) v += __shfl_xor(v, m, 32);
  return v;
}

#define WMMA_BF16(A, Bf, C) \
  __builtin_amdgcn_wmma_f32_16x16x32_bf16(false, (A), false, (Bf), (short)0, (C), false, false)

// ---------------------------------------------------------------- f32 -> bf16
__global__ void cvt_bf16_kernel(const float* __restrict__ src,
                                unsigned short* __restrict__ dst, size_t n) {
  size_t i = (size_t)blockIdx.x * blockDim.x + threadIdx.x;
  size_t stride = (size_t)gridDim.x * blockDim.x;
  for (; i < n; i += stride) dst[i] = f2bf(src[i]);
}

// ----------------------------------------------------- V transpose (per head)
// vT[((b*H+h)*64 + d)*512 + n] = kv[(b*N+n)*2048 + 1024 + h*64 + d]
__global__ void transpose_v_kernel(const unsigned short* __restrict__ kvbf,
                                   unsigned short* __restrict__ vT) {
  const size_t total = (size_t)8 * 16 * 64 * 512;
  size_t i = (size_t)blockIdx.x * blockDim.x + threadIdx.x;
  size_t stride = (size_t)gridDim.x * blockDim.x;
  for (; i < total; i += stride) {
    int n = (int)(i & 511);
    int d = (int)((i >> 9) & 63);
    int h = (int)((i >> 15) & 15);
    int b = (int)(i >> 19);
    vT[i] = kvbf[((size_t)(b * 512 + n)) * 2048 + 1024 + h * 64 + d];
  }
}

// ------------------------------------------------------- generic WMMA GEMM
// C[M,N] = A[M,K] @ W[N,K]^T + bias ; optional relu; f32 and/or bf16 outputs.
// 8 waves: 4 M-rows x 2 N-cols; block tile 64x128, wave tile 16x64.
// Requires K % 64 == 0 and K >= 64 (true for all call sites: 1024 / 4096).
__global__ __launch_bounds__(256) void wmma_gemm_kernel(
    const unsigned short* __restrict__ A, const unsigned short* __restrict__ W,
    const float* __restrict__ bias, float* __restrict__ Cf,
    unsigned short* __restrict__ Cb, int M, int N, int K, int relu) {
  int w    = threadIdx.x >> 5;
  int lane = threadIdx.x & 31;
  int m0 = blockIdx.y * 64 + (w >> 1) * 16;
  int n0 = blockIdx.x * 128 + (w & 1) * 64;

  v8f acc[4] = {{}, {}, {}, {}};
  const unsigned short* Ap  = A + (size_t)m0 * K;
  const unsigned short* Wp0 = W + (size_t)(n0 +  0) * K;
  const unsigned short* Wp1 = W + (size_t)(n0 + 16) * K;
  const unsigned short* Wp2 = W + (size_t)(n0 + 32) * K;
  const unsigned short* Wp3 = W + (size_t)(n0 + 48) * K;

  // stage 0 = k, stage 1 = k+32 (disjoint register sets, no rotation moves)
  v16bf a0  = fragA(Ap, K);
  v16bf p00 = fragB(Wp0, K), p01 = fragB(Wp1, K);
  v16bf p02 = fragB(Wp2, K), p03 = fragB(Wp3, K);
  v16bf a1  = fragA(Ap + 32, K);
  v16bf p10 = fragB(Wp0 + 32, K), p11 = fragB(Wp1 + 32, K);
  v16bf p12 = fragB(Wp2 + 32, K), p13 = fragB(Wp3 + 32, K);

  for (int k = 0; k < K - 64; k += 64) {
    // consume stage 0 (k), prefetch k+64 into stage 0
    acc[0] = WMMA_BF16(a0, p00, acc[0]);
    acc[1] = WMMA_BF16(a0, p01, acc[1]);
    acc[2] = WMMA_BF16(a0, p02, acc[2]);
    acc[3] = WMMA_BF16(a0, p03, acc[3]);
    a0  = fragA(Ap  + k + 64, K);
    p00 = fragB(Wp0 + k + 64, K);
    p01 = fragB(Wp1 + k + 64, K);
    p02 = fragB(Wp2 + k + 64, K);
    p03 = fragB(Wp3 + k + 64, K);
    // consume stage 1 (k+32), prefetch k+96 into stage 1
    acc[0] = WMMA_BF16(a1, p10, acc[0]);
    acc[1] = WMMA_BF16(a1, p11, acc[1]);
    acc[2] = WMMA_BF16(a1, p12, acc[2]);
    acc[3] = WMMA_BF16(a1, p13, acc[3]);
    a1  = fragA(Ap  + k + 96, K);
    p10 = fragB(Wp0 + k + 96, K);
    p11 = fragB(Wp1 + k + 96, K);
    p12 = fragB(Wp2 + k + 96, K);
    p13 = fragB(Wp3 + k + 96, K);
  }
  // epilogue: k = K-64 and K-32 already prefetched
  acc[0] = WMMA_BF16(a0, p00, acc[0]);
  acc[1] = WMMA_BF16(a0, p01, acc[1]);
  acc[2] = WMMA_BF16(a0, p02, acc[2]);
  acc[3] = WMMA_BF16(a0, p03, acc[3]);
  acc[0] = WMMA_BF16(a1, p10, acc[0]);
  acc[1] = WMMA_BF16(a1, p11, acc[1]);
  acc[2] = WMMA_BF16(a1, p12, acc[2]);
  acc[3] = WMMA_BF16(a1, p13, acc[3]);

  int rowoff = (lane >> 4) * 8;
  int c0 = n0 + (lane & 15);
#pragma unroll
  for (int t = 0; t < 4; ++t) {
    int col = c0 + t * 16;
    float bv = bias ? bias[col] : 0.0f;
#pragma unroll
    for (int i = 0; i < 8; ++i) {
      float v = acc[t][i] + bv;
      if (relu) v = v > 0.0f ? v : 0.0f;
      size_t idx = (size_t)(m0 + i + rowoff) * N + col;
      if (Cf) Cf[idx] = v;
      if (Cb) Cb[idx] = f2bf(v);
    }
  }
}

// ------------------------------------------------------------- attention
// grid (P/64, H, B), block 128 (4 waves). Each wave: 16 query rows.
// K tile (512x64 bf16 = 64KB) staged to LDS once per block via TDM.
__global__ __launch_bounds__(128) void attn_kernel(
    const unsigned short* __restrict__ qbf, const unsigned short* __restrict__ kvbf,
    const unsigned short* __restrict__ vT,
    const unsigned char* __restrict__ mask, const float* __restrict__ log_tau,
    float* __restrict__ wout, unsigned short* __restrict__ attnbf) {
  const int Pq = 1024, Nk = 512, D = 1024, H = 16;
  const unsigned KT_OFF = 128u * 1024u;              // LDS byte offset of K tile
  extern __shared__ char smem[];                      // 128KB scores + 64KB K
  float* sS = (float*)smem;
  unsigned short* kT = (unsigned short*)(smem + KT_OFF);

  int w = threadIdx.x >> 5, lane = threadIdx.x & 31;
  int b = blockIdx.z, h = blockIdx.y;
  int p0 = blockIdx.x * 64 + w * 16;
  float scale = 0.125f * __expf(-log_tau[h]);        // 1/sqrt(64)/exp(tau)
  float* Pw = sS + (size_t)w * 16 * 512;

  // ---- TDM: DMA the K tile [512 rows x 64 bf16] for (b,h) into LDS.
  if (threadIdx.x == 0) {
    unsigned long long ga =
        (unsigned long long)(kvbf + ((size_t)b * Nk) * 2048 + h * 64);
    u32x4 g0;
    g0[0] = 1u;                                       // count=1 (valid D#)
    g0[1] = KT_OFF;                                   // lds_addr (bytes)
    g0[2] = (unsigned)ga;                             // global_addr[31:0]
    g0[3] = (unsigned)(ga >> 32) | 0x80000000u;       // global_addr[56:32] | type=2
    i32x8 g1;
    g1[0] = (1 << 16);                                // data_size = 2 bytes
    g1[1] = (64 << 16);                               // tensor_dim0 = 64 (lo16)
    g1[2] = (512 << 16);                              // dim0 hi | tensor_dim1 = 512 (lo16)
    g1[3] = (64 << 16);                               // dim1 hi | tile_dim0 = 64
    g1[4] = 512;                                      // tile_dim1 = 512, tile_dim2 = 0
    g1[5] = 2048;                                     // tensor_dim0_stride = 2048 elems
    g1[6] = 0;
    g1[7] = 0;
    i32x4 z4 = {0, 0, 0, 0};
#if __clang_major__ >= 23
    i32x8 z8 = {0, 0, 0, 0, 0, 0, 0, 0};
    __builtin_amdgcn_tensor_load_to_lds(g0, g1, z4, z4, z8, 0);
#else
    __builtin_amdgcn_tensor_load_to_lds(g0, g1, z4, z4, 0);
#endif
    __builtin_amdgcn_s_wait_tensorcnt(0);
  }
  __syncthreads();

  // ---- phase 1: scores (Q fragments hoisted; K from LDS)
  const unsigned short* Qbase = qbf + ((size_t)(b * Pq + p0)) * D + h * 64;
  v16bf qa0 = fragA(Qbase, D);
  v16bf qa1 = fragA(Qbase + 32, D);
  for (int nt = 0; nt < 32; ++nt) {
    v8f acc = {};
    const unsigned short* Kb = kT + (size_t)(nt * 16) * 64;
    v16bf bb0 = fragB(Kb, 64);
    v16bf bb1 = fragB(Kb + 32, 64);
    acc = WMMA_BF16(qa0, bb0, acc);
    acc = WMMA_BF16(qa1, bb1, acc);
    int rowoff = (lane >> 4) * 8, col = nt * 16 + (lane & 15);
#pragma unroll
    for (int i = 0; i < 8; ++i) Pw[(size_t)(i + rowoff) * 512 + col] = acc[i] * scale;
  }

  // ---- phase 2: softmax over N=512 (per wave, per row; 16 cols per lane)
  for (int r = 0; r < 16; ++r) {
    float* rowp = Pw + (size_t)r * 512;
    float sv[16];
#pragma unroll
    for (int j = 0; j < 16; ++j) {
      int c = lane + j * 32;
      sv[j] = mask[(size_t)b * Nk + c] ? rowp[c] : -__builtin_inff();
    }
    float mx = sv[0];
#pragma unroll
    for (int j = 1; j < 16; ++j) mx = fmaxf(mx, sv[j]);
    mx = wred_max(mx);
    float s = 0.0f;
#pragma unroll
    for (int j = 0; j < 16; ++j) { sv[j] = __expf(sv[j] - mx); s += sv[j]; }
    s = wred_sum(s);
    float inv = s > 0.0f ? 1.0f / s : 0.0f;
    size_t wbase = (((size_t)b * H + h) * Pq + (p0 + r)) * Nk;
#pragma unroll
    for (int j = 0; j < 16; ++j) {
      int c = lane + j * 32;
      float wv = sv[j] * inv;
      rowp[c] = wv;
      wout[wbase + c] = wv;
    }
  }

  // ---- phase 3: attn = probs @ V (K-dim 512, d=64 in 4 tiles; V^T is [64][512])
  v8f oacc[4] = {{}, {}, {}, {}};
  int m = lane & 15, khalf = (lane >> 4) << 3;
  const unsigned short* vTh = vT + ((size_t)(b * H + h) * 64) * 512;
  for (int nc = 0; nc < 16; ++nc) {
    const float* pr = Pw + (size_t)m * 512 + nc * 32;
    v8u ua;
#pragma unroll
    for (int i = 0; i < 4; ++i) {
      ua[i]     = pack2(pr[khalf + 2 * i],      pr[khalf + 2 * i + 1]);
      ua[4 + i] = pack2(pr[khalf + 16 + 2 * i], pr[khalf + 17 + 2 * i]);
    }
    v16bf a = __builtin_bit_cast(v16bf, ua);
#pragma unroll
    for (int t = 0; t < 4; ++t) {
      v16bf bb = fragB(vTh + (size_t)(t * 16) * 512 + nc * 32, 512);
      oacc[t] = WMMA_BF16(a, bb, oacc[t]);
    }
  }
  int rowoff = (lane >> 4) * 8, c0 = h * 64 + (lane & 15);
#pragma unroll
  for (int t = 0; t < 4; ++t)
#pragma unroll
    for (int i = 0; i < 8; ++i)
      attnbf[((size_t)(b * Pq + p0 + i + rowoff)) * D + c0 + t * 16] = f2bf(oacc[t][i]);
}

// ----------------------------------------------- residual + LayerNorm (+cls)
__global__ __launch_bounds__(256) void ln_kernel(
    const float* __restrict__ A, const float* __restrict__ R,
    const float* __restrict__ g, const float* __restrict__ be,
    float* __restrict__ outf, unsigned short* __restrict__ outb,
    const float* __restrict__ clsw, const float* __restrict__ clsb,
    float* __restrict__ logits, float* __restrict__ sig) {
  const int D = 1024;
  int row = blockIdx.x, tid = threadIdx.x;
  __shared__ float red[256];
  __shared__ float sm, svar;
  size_t base = (size_t)row * D;
  float v[4];
  float s = 0.0f;
#pragma unroll
  for (int i = 0; i < 4; ++i) {
    int c = tid + i * 256;
    v[i] = A[base + c] + R[base + c];
    s += v[i];
  }
  red[tid] = s; __syncthreads();
  for (int o = 128; o > 0; o >>= 1) { if (tid < o) red[tid] += red[tid + o]; __syncthreads(); }
  if (tid == 0) sm = red[0] * (1.0f / D);
  __syncthreads();
  float mn = sm;
  s = 0.0f;
#pragma unroll
  for (int i = 0; i < 4; ++i) { float d = v[i] - mn; s += d * d; }
  red[tid] = s; __syncthreads();
  for (int o = 128; o > 0; o >>= 1) { if (tid < o) red[tid] += red[tid + o]; __syncthreads(); }
  if (tid == 0) svar = red[0] * (1.0f / D);
  __syncthreads();
  float inv = rsqrtf(svar + 1e-5f);
  float dot = 0.0f;
#pragma unroll
  for (int i = 0; i < 4; ++i) {
    int c = tid + i * 256;
    float y = (v[i] - mn) * inv * g[c] + be[c];
    if (outf) outf[base + c] = y;
    if (outb) outb[base + c] = f2bf(y);
    if (clsw) dot += y * clsw[c];
  }
  if (clsw) {
    red[tid] = dot; __syncthreads();
    for (int o = 128; o > 0; o >>= 1) { if (tid < o) red[tid] += red[tid + o]; __syncthreads(); }
    if (tid == 0) {
      float L = red[0] + clsb[0];
      logits[row] = L;
      sig[row] = 1.0f / (1.0f + __expf(-L));
    }
  }
}

// --------------------------------------------------------------------- host
extern "C" void kernel_launch(void* const* d_in, const int* in_sizes, int n_in,
                              void* d_out, int out_size, void* d_ws, size_t ws_size,
                              hipStream_t stream) {
  const int B = 8, P = 1024, N = 512, D = 1024, H = 16, HID = 4096;
  const float* img  = (const float*)d_in[0];
  const float* txt  = (const float*)d_in[1];
  const unsigned char* mask = (const unsigned char*)d_in[2];
  const float* win  = (const float*)d_in[3];
  const float* bin  = (const float*)d_in[4];
  const float* wo   = (const float*)d_in[5];
  const float* bo   = (const float*)d_in[6];
  const float* ltau = (const float*)d_in[7];
  const float* g1   = (const float*)d_in[8];
  const float* be1  = (const float*)d_in[9];
  const float* w1   = (const float*)d_in[10];
  const float* b1   = (const float*)d_in[11];
  const float* w2   = (const float*)d_in[12];
  const float* b2   = (const float*)d_in[13];
  const float* g2   = (const float*)d_in[14];
  const float* be2  = (const float*)d_in[15];
  const float* clsw = (const float*)d_in[16];
  const float* clsb = (const float*)d_in[17];

  const size_t MP = (size_t)B * P;   // 8192
  const size_t MN = (size_t)B * N;   // 4096

  char* ws = (char*)d_ws;
  size_t off = 0;
  auto alloc = [&](size_t bytes) -> char* {
    char* p = ws + off;
    off += (bytes + 255) & ~(size_t)255;
    return p;
  };
  unsigned short* img_bf  = (unsigned short*)alloc(MP * D * 2);
  unsigned short* txt_bf  = (unsigned short*)alloc(MN * D * 2);
  unsigned short* win_bf  = (unsigned short*)alloc((size_t)3 * D * D * 2);
  unsigned short* wout_bf = (unsigned short*)alloc((size_t)D * D * 2);
  unsigned short* w1_bf   = (unsigned short*)alloc((size_t)HID * D * 2);
  unsigned short* w2_bf   = (unsigned short*)alloc((size_t)D * HID * 2);
  unsigned short* q_bf    = (unsigned short*)alloc(MP * D * 2);
  unsigned short* kv_bf   = (unsigned short*)alloc(MN * 2 * D * 2);
  unsigned short* vT_bf   = (unsigned short*)alloc((size_t)B * H * 64 * 512 * 2);
  unsigned short* attn_bf = (unsigned short*)alloc(MP * D * 2);
  unsigned short* x1_bf   = (unsigned short*)alloc(MP * D * 2);
  unsigned short* h1_bf   = (unsigned short*)alloc(MP * HID * 2);
  float* proj_f = (float*)alloc(MP * D * 4);
  float* x1_f   = (float*)alloc(MP * D * 4);
  float* ff2_f  = (float*)alloc(MP * D * 4);

  float* out    = (float*)d_out;
  float* x_out  = out;                                   // B,P,D
  float* w_out  = out + MP * (size_t)D;                  // B,H,P,N
  float* logits = w_out + (size_t)B * H * P * N;         // B,P
  float* sigm   = logits + MP;                           // B,P

  // 1) bf16 conversions
  cvt_bf16_kernel<<<2048, 256, 0, stream>>>(img, img_bf, MP * D);
  cvt_bf16_kernel<<<2048, 256, 0, stream>>>(txt, txt_bf, MN * D);
  cvt_bf16_kernel<<<2048, 256, 0, stream>>>(win, win_bf, (size_t)3 * D * D);
  cvt_bf16_kernel<<<1024, 256, 0, stream>>>(wo, wout_bf, (size_t)D * D);
  cvt_bf16_kernel<<<2048, 256, 0, stream>>>(w1, w1_bf, (size_t)HID * D);
  cvt_bf16_kernel<<<2048, 256, 0, stream>>>(w2, w2_bf, (size_t)D * HID);

  dim3 blk(256);
  // 2) q = img @ Wq^T + bq  (bf16 out)
  wmma_gemm_kernel<<<dim3(D / 128, MP / 64), blk, 0, stream>>>(
      img_bf, win_bf, bin, nullptr, q_bf, (int)MP, D, D, 0);
  // 3) kv = txt @ Wkv^T + bkv  (bf16 out)
  wmma_gemm_kernel<<<dim3(2 * D / 128, MN / 64), blk, 0, stream>>>(
      txt_bf, win_bf + (size_t)D * D, bin + D, nullptr, kv_bf, (int)MN, 2 * D, D, 0);
  // 3b) per-head V transpose for clean B-fragment loads
  transpose_v_kernel<<<2048, 256, 0, stream>>>(kv_bf, vT_bf);
  // 4) attention (TDM K-staging; writes f32 weights to d_out, bf16 attn to ws)
  attn_kernel<<<dim3(P / 64, H, B), dim3(128), 128 * 1024 + 64 * 1024, stream>>>(
      q_bf, kv_bf, vT_bf, mask, ltau, w_out, attn_bf);
  // 5) out projection (f32 out)
  wmma_gemm_kernel<<<dim3(D / 128, MP / 64), blk, 0, stream>>>(
      attn_bf, wout_bf, bo, proj_f, nullptr, (int)MP, D, D, 0);
  // 6) x1 = LN(img + proj)
  ln_kernel<<<(int)MP, 256, 0, stream>>>(img, proj_f, g1, be1, x1_f, x1_bf,
                                         nullptr, nullptr, nullptr, nullptr);
  // 7) h1 = relu(x1 @ w1^T + b1)  (bf16 out)
  wmma_gemm_kernel<<<dim3(HID / 128, MP / 64), blk, 0, stream>>>(
      x1_bf, w1_bf, b1, nullptr, h1_bf, (int)MP, HID, D, 1);
  // 8) ff2 = h1 @ w2^T + b2  (f32 out)
  wmma_gemm_kernel<<<dim3(D / 128, MP / 64), blk, 0, stream>>>(
      h1_bf, w2_bf, b2, ff2_f, nullptr, (int)MP, D, HID, 0);
  // 9) x = LN(x1 + ff2); logits = x @ cls_w^T + cls_b; sigmoid
  ln_kernel<<<(int)MP, 256, 0, stream>>>(x1_f, ff2_f, g2, be2, x_out, nullptr,
                                         clsw, clsb, logits, sigm);
}
